// LSTM_73263552135302
// MI455X (gfx1250) — matrix operand order repaired and checked by reading, not compiled, not run on hardware
//
#include <hip/hip_runtime.h>
#include <hip/hip_bf16.h>
#include <math.h>

// Problem constants (from the reference)
#define L 4096
#define E 1024
#define H 1024
#define V 50257

typedef __attribute__((ext_vector_type(16))) _Float16 v16h;
typedef __attribute__((ext_vector_type(8)))  _Float16 v8h;
typedef __attribute__((ext_vector_type(4)))  _Float16 v4h;
typedef __attribute__((ext_vector_type(8)))  float    v8f;

// ---------------------------------------------------------------------------
// f32 -> f16 conversion (vectorized, grid-stride). n must be a multiple of 4.
// ---------------------------------------------------------------------------
__global__ __launch_bounds__(256) void cvt_f32_f16(const float* __restrict__ s,
                                                   _Float16* __restrict__ d,
                                                   long n4) {
    long i = (long)blockIdx.x * blockDim.x + threadIdx.x;
    long stride = (long)gridDim.x * blockDim.x;
    for (; i < n4; i += stride) {
        float4 v = ((const float4*)s)[i];
        v4h o;
        o[0] = (_Float16)v.x; o[1] = (_Float16)v.y;
        o[2] = (_Float16)v.z; o[3] = (_Float16)v.w;
        ((v4h*)d)[i] = o;
    }
}

// ---------------------------------------------------------------------------
// Embedding gather + f16 conversion: x16[l,:] = f16(emb[sentence[l],:])
// One block per token, 256 threads * 4 floats = 1024 = E.
// ---------------------------------------------------------------------------
__global__ __launch_bounds__(256) void gather_embed(const float* __restrict__ emb,
                                                    const int* __restrict__ sent,
                                                    _Float16* __restrict__ x16) {
    int l = blockIdx.x;
    int t = threadIdx.x;
    int tok = sent[l];
    float4 v = ((const float4*)(emb + (size_t)tok * E))[t];
    v4h o;
    o[0] = (_Float16)v.x; o[1] = (_Float16)v.y;
    o[2] = (_Float16)v.z; o[3] = (_Float16)v.w;
    ((v4h*)(x16 + (size_t)l * E))[t] = o;
}

// ---------------------------------------------------------------------------
// WMMA GEMM: C[M,N] = A[M,K] * W[N,K]^T (+ bias[N]), f16 in / f32 out.
// Wave32; each wave computes a 64x64 supertile: 4 A-frags x 4 B-frags,
// 16 accumulators, K-loop in steps of 32 (V_WMMA_F32_16X16X32_F16).
// M % 64 == 0 and K % 32 == 0 assumed; ragged N handled WITHOUT inner-loop
// divergence: out-of-range columns are clamped to N-1 for loads (valid
// memory, garbage accumulators) and only the final stores are masked.
// ---------------------------------------------------------------------------
#define GEMM_TM 4
#define GEMM_TN 4
#define WAVES_PER_BLOCK 8

__global__ __launch_bounds__(256) void wmma_gemm_f16f32(
    const _Float16* __restrict__ A,
    const _Float16* __restrict__ W,
    float* __restrict__ C,
    int M, int N, int K,
    const float* __restrict__ bias,
    int tilesN) {
    const int lane = threadIdx.x & 31;
    const int wave = threadIdx.x >> 5;
    const int waveId = blockIdx.x * WAVES_PER_BLOCK + wave;
    const int tilesM = M >> 6;
    if (waveId >= tilesM * tilesN) return;   // wave-uniform exit (EXEC stays full)

    const int tm = waveId / tilesN;
    const int tn = waveId % tilesN;
    const int m0 = tm << 6;                  // 64 rows
    const int n0 = tn << 6;                  // 64 cols

    const int half = lane >> 4;              // lane-group within the wave
    const int l15  = lane & 15;

    // Loop-invariant base pointers.
    const _Float16* aBase[GEMM_TM];
#pragma unroll
    for (int i = 0; i < GEMM_TM; ++i)
        aBase[i] = A + (size_t)(m0 + 16 * i + l15) * K + half * 16;

    int colIdx[GEMM_TN];
    const _Float16* bBase[GEMM_TN];
#pragma unroll
    for (int j = 0; j < GEMM_TN; ++j) {
        int col = n0 + 16 * j + l15;
        colIdx[j] = col;
        int colc = col < N ? col : (N - 1);          // clamp: keeps EXEC uniform
        bBase[j] = W + (size_t)colc * K + half * 16;
    }

    v8f acc[GEMM_TM][GEMM_TN];
#pragma unroll
    for (int i = 0; i < GEMM_TM; ++i)
#pragma unroll
        for (int j = 0; j < GEMM_TN; ++j)
            acc[i][j] = (v8f){};

    for (int k0 = 0; k0 < K; k0 += 32) {
        // ---- A fragments: row m0+16i+l15, K range 16h..16h+15 (contiguous)
        v16h a[GEMM_TM];
#pragma unroll
        for (int i = 0; i < GEMM_TM; ++i) {
            v8h lo = *(const v8h*)(aBase[i] + k0);
            v8h hi = *(const v8h*)(aBase[i] + k0 + 8);
#pragma unroll
            for (int q = 0; q < 8; ++q) { a[i][q] = lo[q]; a[i][8 + q] = hi[q]; }
        }
        // ---- B fragments: column n0+16j+l15 is a K-contiguous row of W
        v16h b[GEMM_TN];
#pragma unroll
        for (int j = 0; j < GEMM_TN; ++j)
            b[j] = *(const v16h*)(bBase[j] + k0);

        // ---- 16 WMMAs, full EXEC, no branches
#pragma unroll
        for (int i = 0; i < GEMM_TM; ++i)
#pragma unroll
            for (int j = 0; j < GEMM_TN; ++j)
                acc[i][j] = __builtin_amdgcn_wmma_f32_16x16x32_f16(
                    false, a[i], false, b[j], (short)0, acc[i][j], false, false);
    }

    // ---- Store: VGPR r <-> row (half*8 + r), col = l15 within the 16x16 tile
#pragma unroll
    for (int j = 0; j < GEMM_TN; ++j) {
        int col = colIdx[j];
        if (col < N) {
            float bv = bias ? bias[col] : 0.0f;
#pragma unroll
            for (int i = 0; i < GEMM_TM; ++i) {
                int rowb = m0 + 16 * i + half * 8;
#pragma unroll
                for (int r = 0; r < 8; ++r)
                    C[(size_t)(rowb + r) * N + col] = acc[i][j][r] + bv;
            }
        }
    }
}

// ---------------------------------------------------------------------------
// LSTM cell elementwise (c0 = h0 = 0, so the f-gate contributes nothing):
//   i = gates[:, j], g = gates[:, 2048+j], o = gates[:, 3072+j]  (+ biases)
//   c = sigmoid(i)*tanh(g);  h = sigmoid(o)*tanh(c)  -> f16 for the big GEMM
// ---------------------------------------------------------------------------
__global__ __launch_bounds__(256) void lstm_act(const float* __restrict__ gates,
                                                const float* __restrict__ b_ih,
                                                const float* __restrict__ b_hh,
                                                _Float16* __restrict__ h16) {
    int idx = blockIdx.x * blockDim.x + threadIdx.x;   // L*H threads
    int l = idx >> 10;
    int j = idx & 1023;
    const float* g = gates + (size_t)l * (4 * H);
    float iv = g[j]        + b_ih[j]        + b_hh[j];
    float gv = g[2048 + j] + b_ih[2048 + j] + b_hh[2048 + j];
    float ov = g[3072 + j] + b_ih[3072 + j] + b_hh[3072 + j];
    float si = 1.0f / (1.0f + __expf(-iv));
    float so = 1.0f / (1.0f + __expf(-ov));
    float c  = si * tanhf(gv);
    float hv = so * tanhf(c);
    h16[idx] = (_Float16)hv;
}

// ---------------------------------------------------------------------------
// Row-wise log-softmax over N=V, in place. One block (256 threads) per row.
// Online max/sum per thread, then LDS tree combine, then subtract pass.
// ---------------------------------------------------------------------------
__global__ __launch_bounds__(256) void logsoftmax_rows(float* __restrict__ out, int N) {
    __shared__ float sm[256];
    __shared__ float ss[256];
    float* row = out + (size_t)blockIdx.x * N;
    float m = -INFINITY, s = 0.0f;
    for (int i = threadIdx.x; i < N; i += 256) {
        float x = row[i];
        if (x > m) { s = s * __expf(m - x) + 1.0f; m = x; }
        else       { s += __expf(x - m); }
    }
    sm[threadIdx.x] = m;
    ss[threadIdx.x] = s;
    __syncthreads();
    for (int off = 128; off > 0; off >>= 1) {
        if (threadIdx.x < off) {
            float m1 = sm[threadIdx.x], s1 = ss[threadIdx.x];
            float m2 = sm[threadIdx.x + off], s2 = ss[threadIdx.x + off];
            float M = fmaxf(m1, m2);
            sm[threadIdx.x] = M;
            ss[threadIdx.x] = s1 * __expf(m1 - M) + s2 * __expf(m2 - M);
        }
        __syncthreads();
    }
    float lse = sm[0] + __logf(ss[0]);
    for (int i = threadIdx.x; i < N; i += 256)
        row[i] -= lse;
}

// ---------------------------------------------------------------------------
// Orchestration
// ---------------------------------------------------------------------------
extern "C" void kernel_launch(void* const* d_in, const int* in_sizes, int n_in,
                              void* d_out, int out_size, void* d_ws, size_t ws_size,
                              hipStream_t stream) {
    const int*   sentence = (const int*)  d_in[0];
    const float* emb      = (const float*)d_in[1];
    const float* W_ih     = (const float*)d_in[2];
    // d_in[3] = W_hh: unused, h0 == 0 so h0 @ W_hh^T == 0 exactly
    const float* b_ih     = (const float*)d_in[4];
    const float* b_hh     = (const float*)d_in[5];
    const float* W_out    = (const float*)d_in[6];
    const float* b_out    = (const float*)d_in[7];
    float* out = (float*)d_out;

    // Workspace carve (all offsets 256B aligned):
    //   W_out f16 (103 MB, fits in the 192 MB L2), W_ih f16, x f16, h f16.
    char* ws = (char*)d_ws;
    const size_t szWout = (size_t)V * H * sizeof(_Float16);       // 102,926,336
    const size_t szWih  = (size_t)4 * H * E * sizeof(_Float16);   //   8,388,608
    const size_t szX    = (size_t)L * E * sizeof(_Float16);       //   8,388,608
    _Float16* Wout16 = (_Float16*)(ws);
    _Float16* Wih16  = (_Float16*)(ws + szWout);
    _Float16* x16    = (_Float16*)(ws + szWout + szWih);
    _Float16* h16    = (_Float16*)(ws + szWout + szWih + szX);

    // Gates [L, 4H] f32 staged inside d_out (64 MB), overwritten later by logits.
    float* gates = (float*)d_out;

    // 1) Weight conversions to f16
    cvt_f32_f16<<<8192, 256, 0, stream>>>(W_out, Wout16, (long)V * H / 4);
    cvt_f32_f16<<<4096, 256, 0, stream>>>(W_ih, Wih16, (long)4 * H * E / 4);

    // 2) Embedding gather + f16 convert
    gather_embed<<<L, 256, 0, stream>>>(emb, sentence, x16);

    // 3) gates = x @ W_ih^T   (M=4096, N=4096, K=1024)
    {
        int tilesN = (4 * H) / 64;                       // 64
        int waves  = (L / 64) * tilesN;                  // 4096
        int blocks = (waves + WAVES_PER_BLOCK - 1) / WAVES_PER_BLOCK;
        wmma_gemm_f16f32<<<blocks, 256, 0, stream>>>(
            x16, Wih16, gates, L, 4 * H, E, nullptr, tilesN);
    }

    // 4) LSTM cell elementwise -> h16
    lstm_act<<<(L * H) / 256, 256, 0, stream>>>(gates, b_ih, b_hh, h16);

    // 5) logits = h @ W_out^T + b_out   (M=4096, N=50257, K=1024)
    {
        int tilesN = (V + 63) / 64;                      // 786 (ragged tail)
        int waves  = (L / 64) * tilesN;                  // 50,304
        int blocks = (waves + WAVES_PER_BLOCK - 1) / WAVES_PER_BLOCK;
        wmma_gemm_f16f32<<<blocks, 256, 0, stream>>>(
            h16, Wout16, out, L, V, H, b_out, tilesN);
    }

    // 6) log_softmax rows in place
    logsoftmax_rows<<<L, 256, 0, stream>>>(out, V);
}
